// SupConLoss_86088324481780
// MI455X (gfx1250) — matrix operand order repaired and checked by reading, not compiled, not run on hardware
//
#include <hip/hip_runtime.h>
#include <hip/hip_bf16.h>
#include <math.h>

typedef __attribute__((ext_vector_type(16))) _Float16 v16h;
typedef __attribute__((ext_vector_type(8)))  _Float16 v8h;
typedef __attribute__((ext_vector_type(8)))  float    v8f;

#define B_ROWS   8192
#define DIMS     256
#define NSPLIT   8            // column-split for occupancy
#define TILE_N   64           // columns staged in LDS per iteration
#define KT       (DIMS / 32)  // 8 WMMA K-steps of 32
#define LDS_PITCH (DIMS + 8)  // +16B/row padding -> conflict-free b128 reads
#define NTILES   ((B_ROWS / NSPLIT) / TILE_N)

// ---------------------------------------------------------------------------
// Kernel 1: L2-normalize rows (f32 -> f16).  One wave (32 lanes) per row.
// ---------------------------------------------------------------------------
__global__ void __launch_bounds__(256) normalize_f16_kernel(
    const float* __restrict__ feats, _Float16* __restrict__ out)
{
    const int wave = threadIdx.x >> 5;
    const int lane = threadIdx.x & 31;
    const int row  = blockIdx.x * 8 + wave;
    if (row >= B_ROWS) return;

    const float* src = feats + (size_t)row * DIMS;
    float x[DIMS / 32];
    float ss = 0.f;
#pragma unroll
    for (int i = 0; i < DIMS / 32; ++i) {
        x[i] = src[lane + 32 * i];
        ss += x[i] * x[i];
    }
#pragma unroll
    for (int m = 16; m >= 1; m >>= 1) ss += __shfl_xor(ss, m, 32);

    const float inv = 1.0f / fmaxf(sqrtf(ss), 1e-12f);
    _Float16* dst = out + (size_t)row * DIMS;
#pragma unroll
    for (int i = 0; i < DIMS / 32; ++i)
        dst[lane + 32 * i] = (_Float16)(x[i] * inv);
}

// ---------------------------------------------------------------------------
// Kernel 2: fused  exp(F F^T / T)  masked row-sums via v_wmma_f32_16x16x32_f16
// with double-buffered GLOBAL_LOAD_ASYNC_TO_LDS_B128 staging of B columns.
// Grid: (64 row-blocks, NSPLIT column splits), 256 threads = 8 waves.
// ---------------------------------------------------------------------------
__global__ void __launch_bounds__(256) supcon_fused_kernel(
    const _Float16* __restrict__ F16, const int* __restrict__ labels,
    float* __restrict__ part_num, float* __restrict__ part_den)
{
    __shared__ __align__(16) _Float16 ldsB[2][TILE_N][LDS_PITCH];

    const int tid  = threadIdx.x;
    const int wave = tid >> 5;
    const int lane = tid & 31;
    const int lh   = lane >> 4;   // lane-half: 0 = lanes 0-15, 1 = lanes 16-31
    const int l16  = lane & 15;

    const int row_base = blockIdx.x * 128 + wave * 16;
    const int col_lo   = blockIdx.y * (B_ROWS / NSPLIT);

    // ---- preload A (16 rows x 256 K) in WMMA layout -----------------------
    // 16-bit A 16x32: lanes 0-15 hold K {0..7,16..23}, lanes 16-31 {8..15,24..31}
    v16h a[KT];
    {
        const _Float16* Arow = F16 + (size_t)(row_base + l16) * DIMS + 8 * lh;
#pragma unroll
        for (int kt = 0; kt < KT; ++kt) {
            v8h lo = *(const v8h*)(Arow + 32 * kt);
            v8h hi = *(const v8h*)(Arow + 32 * kt + 16);
            a[kt] = __builtin_shufflevector(lo, hi, 0, 1, 2, 3, 4, 5, 6, 7,
                                            8, 9, 10, 11, 12, 13, 14, 15);
        }
    }
    int lab_row[8];
#pragma unroll
    for (int v = 0; v < 8; ++v) lab_row[v] = labels[row_base + 8 * lh + v];

    float num[8], den[8];
#pragma unroll
    for (int v = 0; v < 8; ++v) { num[v] = 0.f; den[v] = 0.f; }

    // Async stage of one 64-column tile: 32 KB, 8 x b128 per thread,
    // direct global->LDS (no VGPR round-trip), tracked on ASYNCcnt.
    auto stage = [&](int buf, int tile) {
        const int col_base = col_lo + tile * TILE_N;
#pragma unroll
        for (int i = 0; i < 8; ++i) {
            const int idx = tid + i * 256;  // 0..2047 16-byte chunks
            const int n   = idx >> 5;       // column within tile
            const int ck  = idx & 31;       // 16B chunk within column
            const unsigned ldsOff =
                (unsigned)(uintptr_t)(&ldsB[buf][n][ck * 8]);
            const unsigned long long gaddr = (unsigned long long)(uintptr_t)(
                F16 + (size_t)(col_base + n) * DIMS + ck * 8);
            asm volatile("global_load_async_to_lds_b128 %0, %1, off"
                         :: "v"(ldsOff), "v"(gaddr) : "memory");
        }
    };

#if __has_builtin(__builtin_amdgcn_exp2f)
    const float kExpScale = (float)(1.0 / (0.07 * 0.69314718055994530942));
#else
    const float kInvT = 1.0f / 0.07f;
#endif

    stage(0, 0);  // prologue prefetch

    for (int t = 0; t < NTILES; ++t) {
        const int cur = t & 1;
        // Prefetch next tile into the other buffer (wraps on last iter so the
        // outstanding-async count stays uniform; the extra load is harmless).
        stage(cur ^ 1, (t + 1) % NTILES);
        // Async loads complete in order: <=8 outstanding => tile t's 8 landed.
        asm volatile("s_wait_asynccnt 0x8" ::: "memory");
        __syncthreads();  // all waves' chunks of tile t visible

        const int col_base = col_lo + t * TILE_N;
#pragma unroll
        for (int sub = 0; sub < TILE_N / 16; ++sub) {
            const int gcol = col_base + sub * 16 + l16;
            const int clab = labels[gcol];

            v8f c = {};
#pragma unroll
            for (int kt = 0; kt < KT; ++kt) {
                // 16-bit B 32x16: lanes 0-15 N=l16,K={0..7,16..23}; lanes 16-31 K={8..15,24..31}
                const _Float16* bp =
                    &ldsB[cur][sub * 16 + l16][kt * 32 + 8 * lh];
                v8h lo = *(const v8h*)bp;
                v8h hi = *(const v8h*)(bp + 16);
                v16h b = __builtin_shufflevector(lo, hi, 0, 1, 2, 3, 4, 5, 6, 7,
                                                 8, 9, 10, 11, 12, 13, 14, 15);
                c = __builtin_amdgcn_wmma_f32_16x16x32_f16(
                        false, a[kt], false, b, (short)0, c, false, false);
            }

            // C layout: element v of lane -> (M = v + 8*lh, N = l16)
#pragma unroll
            for (int v = 0; v < 8; ++v) {
                const int grow = row_base + 8 * lh + v;
#if __has_builtin(__builtin_amdgcn_exp2f)
                const float e = __builtin_amdgcn_exp2f(c[v] * kExpScale);
#else
                const float e = __expf(c[v] * kInvT);
#endif
                const float ed = (grow == gcol) ? 0.f : e;
                den[v] += ed;
                num[v] += (clab == lab_row[v]) ? ed : 0.f;
            }
        }
        __syncthreads();  // everyone done reading buf `cur` before overwrite
    }

    // ---- reduce across the 16 lanes holding each row's N-partials ---------
#pragma unroll
    for (int v = 0; v < 8; ++v) {
        float n = num[v], d = den[v];
#pragma unroll
        for (int m = 8; m >= 1; m >>= 1) {
            n += __shfl_xor(n, m, 32);
            d += __shfl_xor(d, m, 32);
        }
        if (l16 == 0) {
            const int r = row_base + 8 * lh + v;
            part_num[blockIdx.y * B_ROWS + r] = n;
            part_den[blockIdx.y * B_ROWS + r] = d;
        }
    }
}

// ---------------------------------------------------------------------------
// Kernel 3: combine splits, loss = -mean(log(num/den)) over rows with num>0.
// ---------------------------------------------------------------------------
__global__ void __launch_bounds__(256) finalize_kernel(
    const float* __restrict__ part_num, const float* __restrict__ part_den,
    float* __restrict__ out)
{
    __shared__ float s_sum[256];
    __shared__ int   s_cnt[256];

    float sum = 0.f;
    int   cnt = 0;
    for (int r = threadIdx.x; r < B_ROWS; r += 256) {
        float n = 0.f, d = 0.f;
#pragma unroll
        for (int s = 0; s < NSPLIT; ++s) {
            n += part_num[s * B_ROWS + r];
            d += part_den[s * B_ROWS + r];
        }
        if (n > 0.f) { sum += logf(n / d); ++cnt; }
    }
    s_sum[threadIdx.x] = sum;
    s_cnt[threadIdx.x] = cnt;
    __syncthreads();
    for (int st = 128; st > 0; st >>= 1) {
        if (threadIdx.x < st) {
            s_sum[threadIdx.x] += s_sum[threadIdx.x + st];
            s_cnt[threadIdx.x] += s_cnt[threadIdx.x + st];
        }
        __syncthreads();
    }
    if (threadIdx.x == 0)
        out[0] = (s_cnt[0] > 0) ? (-s_sum[0] / (float)s_cnt[0]) : 0.f;
}

// ---------------------------------------------------------------------------
extern "C" void kernel_launch(void* const* d_in, const int* in_sizes, int n_in,
                              void* d_out, int out_size, void* d_ws, size_t ws_size,
                              hipStream_t stream) {
    const float* feats  = (const float*)d_in[0];
    const int*   labels = (const int*)d_in[1];
    float*       out    = (float*)d_out;

    _Float16* f16      = (_Float16*)d_ws;                       // 4 MB
    float*    part_num = (float*)((char*)d_ws + (size_t)B_ROWS * DIMS * 2);
    float*    part_den = part_num + NSPLIT * B_ROWS;            // +256 KB each

    normalize_f16_kernel<<<B_ROWS / 8, 256, 0, stream>>>(feats, f16);
    supcon_fused_kernel<<<dim3(B_ROWS / 128, NSPLIT), 256, 0, stream>>>(
        f16, labels, part_num, part_den);
    finalize_kernel<<<1, 256, 0, stream>>>(part_num, part_den, out);
}
